// Net_38233798869763
// MI455X (gfx1250) — matrix operand order repaired and checked by reading, not compiled, not run on hardware
//
#include <hip/hip_runtime.h>
#include <math.h>

// ---------------------------------------------------------------------------
// Types for CDNA5 WMMA (wave32): v_wmma_f32_16x16x32_f16
// ---------------------------------------------------------------------------
typedef __attribute__((ext_vector_type(16))) _Float16 v16h;
typedef __attribute__((ext_vector_type(8)))  float    v8f;
typedef __attribute__((ext_vector_type(4)))  float    f4v;

template<int ACT>
__device__ __forceinline__ float act_apply(float v) {
    if constexpr (ACT == 1) {          // tanh
        return tanhf(v);
    } else if constexpr (ACT == 2) {   // softplus
        return (v > 20.f) ? v : log1pf(expf(v));
    } else {
        return v;
    }
}

__device__ __forceinline__ void cvt_store4(_Float16* dst, f4v d) {
    dst[0] = (_Float16)d.x; dst[1] = (_Float16)d.y;
    dst[2] = (_Float16)d.z; dst[3] = (_Float16)d.w;
}

// ---------------------------------------------------------------------------
// Pipelined WMMA GEMM:  C[M,N] = act(A[M,K] @ W[N,K]^T + bias) (+ res)
//   - 256 threads = 8 waves; 128x128 output tile per workgroup
//   - each wave: 64 rows x 32 cols = 4x2 accumulators of 16x16
//   - double-buffered LDS (f16), register-staged global loads for tile t+1
//     issued before the WMMAs of tile t; global_prefetch warms L2 for t+2
//   - K must be a multiple of 32; M a multiple of 128; N guarded
// ---------------------------------------------------------------------------
template<int ACT>
__global__ __launch_bounds__(256)
void wmma_gemm_kernel(const float* __restrict__ A, int lda,
                      const float* __restrict__ W,      // N x K, row-major
                      const float* __restrict__ bias,   // may be null
                      const float* __restrict__ res,    // may be null (residual add)
                      float* __restrict__ C,
                      int N, int K)
{
    __shared__ _Float16 As[2][128][32];
    __shared__ _Float16 Ws[2][128][32];

    const int tid  = threadIdx.x;
    const int lane = tid & 31;
    const int wave = tid >> 5;
    const int wm   = (wave & 1) * 64;     // wave row offset inside tile
    const int wn   = (wave >> 1) * 32;    // wave col offset inside tile
    const int bm   = blockIdx.y * 128;
    const int bn   = blockIdx.x * 128;

    v8f acc[4][2];
#pragma unroll
    for (int i = 0; i < 4; ++i)
#pragma unroll
        for (int j = 0; j < 2; ++j)
#pragma unroll
            for (int r = 0; r < 8; ++r) acc[i][j][r] = 0.f;

    // cooperative tile loader: thread t handles row t/2, k-halves (t&1)*16
    const int lr = tid >> 1;
    const int lc = (tid & 1) * 16;
    const float* aptr = A + (size_t)(bm + lr) * lda + lc;
    const int    wrow = bn + lr;
    const float* wptr = W + (size_t)wrow * K + lc;
    const bool   wok  = (wrow < N);

    // fragment gather coordinates per ISA VGPR layouts (wave32):
    //  A 16x32 f16: lane<16 -> K {0..7,16..23}; lane>=16 -> K {8..15,24..31}
    //  B 32x16 f16: n = lane&15, K same as A: lane<16 -> {0..7,16..23}; lane>=16 -> {8..15,24..31}
    const int am   = lane & 15;
    const int kh   = (lane >> 4) * 8;
    const int bn16 = lane & 15;
    const int kg   = (lane >> 4) * 8;

    const int nk = K >> 5;  // K / 32 tiles

    // ---- prologue: stage tile 0 into LDS buffer 0
    {
        f4v areg[4], wreg[4];
#pragma unroll
        for (int v = 0; v < 4; ++v) areg[v] = *(const f4v*)(aptr + v * 4);
        if (wok) {
#pragma unroll
            for (int v = 0; v < 4; ++v) wreg[v] = *(const f4v*)(wptr + v * 4);
        } else {
#pragma unroll
            for (int v = 0; v < 4; ++v) wreg[v] = (f4v){0.f, 0.f, 0.f, 0.f};
        }
#pragma unroll
        for (int v = 0; v < 4; ++v) {
            cvt_store4(&As[0][lr][lc + v * 4], areg[v]);
            cvt_store4(&Ws[0][lr][lc + v * 4], wreg[v]);
        }
    }

    for (int t = 0; t < nk; ++t) {
        __syncthreads();
        const int cur = t & 1;

        // ---- stage tile t+1 into registers (latency overlapped with WMMAs)
        f4v areg[4], wreg[4];
        const bool more = (t + 1 < nk);
        if (more) {
            const float* ap = aptr + (t + 1) * 32;
            const float* wp = wptr + (t + 1) * 32;
#pragma unroll
            for (int v = 0; v < 4; ++v) areg[v] = *(const f4v*)(ap + v * 4);
            if (wok) {
#pragma unroll
                for (int v = 0; v < 4; ++v) wreg[v] = *(const f4v*)(wp + v * 4);
            } else {
#pragma unroll
                for (int v = 0; v < 4; ++v) wreg[v] = (f4v){0.f, 0.f, 0.f, 0.f};
            }
            // L2 warm for tile t+2 (gfx1250 global_prefetch_b8)
            if (t + 2 < nk) {
                __builtin_prefetch(aptr + (t + 2) * 32, 0, 1);
                if (wok) __builtin_prefetch(wptr + (t + 2) * 32, 0, 1);
            }
        }

        // ---- compute tile t from LDS buffer `cur`
        v16h bfrag[2];
#pragma unroll
        for (int j = 0; j < 2; ++j) {
            const _Float16* bp = &Ws[cur][wn + j * 16 + bn16][kg];
            ((f4v*)&bfrag[j])[0] = *(const f4v*)(bp);
            ((f4v*)&bfrag[j])[1] = *(const f4v*)(bp + 16);
        }
#pragma unroll
        for (int i = 0; i < 4; ++i) {
            v16h afrag;
            const _Float16* ap = &As[cur][wm + i * 16 + am][0];
            ((f4v*)&afrag)[0] = *(const f4v*)(ap + kh);
            ((f4v*)&afrag)[1] = *(const f4v*)(ap + 16 + kh);
#pragma unroll
            for (int j = 0; j < 2; ++j) {
                acc[i][j] = __builtin_amdgcn_wmma_f32_16x16x32_f16(
                    false, afrag, false, bfrag[j],
                    (short)0, acc[i][j], false, false);
            }
        }

        // ---- drain staged registers into the other LDS buffer
        if (more) {
#pragma unroll
            for (int v = 0; v < 4; ++v) {
                cvt_store4(&As[cur ^ 1][lr][lc + v * 4], areg[v]);
                cvt_store4(&Ws[cur ^ 1][lr][lc + v * 4], wreg[v]);
            }
        }
    }

    // Epilogue. C/D layout: VGPR r, lane -> m = r + 8*(lane>=16), n = lane&15
    const int ml = (lane >> 4) * 8;
    const int nl = lane & 15;
#pragma unroll
    for (int i = 0; i < 4; ++i) {
#pragma unroll
        for (int j = 0; j < 2; ++j) {
            const int ncol = bn + wn + j * 16 + nl;
            if (ncol >= N) continue;
            const float bv = bias ? bias[ncol] : 0.f;
#pragma unroll
            for (int r = 0; r < 8; ++r) {
                const int mrow = bm + wm + i * 16 + ml + r;
                float v = acc[i][j][r] + bv;
                v = act_apply<ACT>(v);
                const size_t idx = (size_t)mrow * N + ncol;
                if (res) v += res[idx];
                C[idx] = v;
            }
        }
    }
}

// ---------------------------------------------------------------------------
// RMSNorm over last dim (one block per row)
// ---------------------------------------------------------------------------
__global__ __launch_bounds__(256)
void rmsnorm_kernel(const float* __restrict__ x, const float* __restrict__ w,
                    float* __restrict__ out, int D)
{
    __shared__ float red[8];
    const int row = blockIdx.x;
    const float* xr = x + (size_t)row * D;
    float s = 0.f;
    for (int i = threadIdx.x; i < D; i += 256) { float v = xr[i]; s += v * v; }
    for (int off = 16; off > 0; off >>= 1) s += __shfl_down(s, off, 32);
    if ((threadIdx.x & 31) == 0) red[threadIdx.x >> 5] = s;
    __syncthreads();
    if (threadIdx.x == 0) {
        float t = 0.f;
        for (int i = 0; i < 8; ++i) t += red[i];
        red[0] = rsqrtf(t / (float)D + 1e-5f);
    }
    __syncthreads();
    const float r = red[0];
    for (int i = threadIdx.x; i < D; i += 256)
        out[(size_t)row * D + i] = xr[i] * r * w[i];
}

// ---------------------------------------------------------------------------
// Seq-len-1 causal depthwise conv (only last tap survives) + SiLU
//   xs[b,d] = silu(xz[b, d] * conv_w[d,3] + conv_b[d])
// ---------------------------------------------------------------------------
__global__ __launch_bounds__(256)
void convsilu_kernel(const float* __restrict__ xz, const float* __restrict__ cw,
                     const float* __restrict__ cb, float* __restrict__ xs,
                     int DI, int total)
{
    const int idx = blockIdx.x * 256 + threadIdx.x;
    if (idx >= total) return;
    const int b = idx / DI, d = idx - b * DI;
    const float v = xz[(size_t)b * 2 * DI + d] * cw[d * 4 + 3] + cb[d];
    xs[idx] = v / (1.f + expf(-v));
}

// ---------------------------------------------------------------------------
// SSM collapse (seq len 1, h0 = 0  =>  y = (delta*dot(B,C) + D_ssm)*xs * silu(z))
// One block per batch row; writes y in place over xs.
// ---------------------------------------------------------------------------
__global__ __launch_bounds__(256)
void combine_kernel(const float* __restrict__ dbl,   // B x 96: [dlo | Bm | Cm]
                    const float* __restrict__ delta, // B x DI (softplus applied)
                    const float* __restrict__ xz,    // B x 2*DI (z in 2nd half)
                    const float* __restrict__ Dssm,  // DI
                    float* __restrict__ xs,          // in: silu(conv), out: y
                    int DI)
{
    __shared__ float sbc;
    const int b = blockIdx.x;
    if (threadIdx.x < 32) {
        float p = 0.f;
        if (threadIdx.x < 16) {
            const float bm = dbl[(size_t)b * 96 + 64 + threadIdx.x];
            const float cm = dbl[(size_t)b * 96 + 80 + threadIdx.x];
            p = bm * cm;
        }
        for (int off = 8; off > 0; off >>= 1) p += __shfl_down(p, off, 32);
        if (threadIdx.x == 0) sbc = p;
    }
    __syncthreads();
    const float bc = sbc;
    for (int d = threadIdx.x; d < DI; d += 256) {
        const size_t i = (size_t)b * DI + d;
        const float xv = xs[i];
        float y = (delta[i] * bc + Dssm[d]) * xv;
        const float z = xz[(size_t)b * 2 * DI + DI + d];
        y *= z / (1.f + expf(-z));
        xs[i] = y;
    }
}

// ---------------------------------------------------------------------------
// Host-side dispatch
// ---------------------------------------------------------------------------
static void launch_gemm(hipStream_t s, int act,
                        const float* A, int lda, const float* W,
                        const float* bias, const float* res, float* C,
                        int M, int N, int K)
{
    dim3 grid((N + 127) / 128, M / 128);
    dim3 blk(256);
    switch (act) {
        case 1: wmma_gemm_kernel<1><<<grid, blk, 0, s>>>(A, lda, W, bias, res, C, N, K); break;
        case 2: wmma_gemm_kernel<2><<<grid, blk, 0, s>>>(A, lda, W, bias, res, C, N, K); break;
        default: wmma_gemm_kernel<0><<<grid, blk, 0, s>>>(A, lda, W, bias, res, C, N, K); break;
    }
}

extern "C" void kernel_launch(void* const* d_in, const int* in_sizes, int n_in,
                              void* d_out, int out_size, void* d_ws, size_t ws_size,
                              hipStream_t stream)
{
    (void)in_sizes; (void)n_in; (void)out_size; (void)ws_size;
    const int Bn = 4096, IN = 512, D = 1024, OUT = 256;
    const int NL = 4, DI = 2048, DTR = 64;

    const float* x         = (const float*)d_in[0];
    const float* pw1       = (const float*)d_in[1];
    const float* pb1       = (const float*)d_in[2];
    const float* pw2       = (const float*)d_in[3];
    const float* pb2       = (const float*)d_in[4];
    const float* norm_w    = (const float*)d_in[5];
    const float* in_proj_w = (const float*)d_in[6];
    const float* conv_w    = (const float*)d_in[7];
    const float* conv_b    = (const float*)d_in[8];
    const float* x_proj_w  = (const float*)d_in[9];
    const float* dt_w      = (const float*)d_in[10];
    const float* dt_b      = (const float*)d_in[11];
    // d_in[12] = A_log: provably unused (seq len 1, h0 == 0 => dA*h0 vanishes)
    const float* D_ssm     = (const float*)d_in[13];
    const float* out_w     = (const float*)d_in[14];
    const float* dw1       = (const float*)d_in[15];
    const float* db1       = (const float*)d_in[16];
    const float* dw2       = (const float*)d_in[17];
    const float* db2       = (const float*)d_in[18];

    // Workspace arena (~146 MB)
    char* ws = (char*)d_ws;
    size_t off = 0;
    float* h    = (float*)(ws + off); off += (size_t)Bn * D * 4;        // 16 MB
    float* xzb  = (float*)(ws + off); off += (size_t)Bn * 2 * DI * 4;   // 64 MB (also liquid tmp / head tmp)
    float* xsb  = (float*)(ws + off); off += (size_t)Bn * DI * 4;       // 32 MB (y written in place)
    float* dltb = (float*)(ws + off); off += (size_t)Bn * DI * 4;       // 32 MB (xn aliases here)
    float* dblb = (float*)(ws + off); off += (size_t)Bn * 96 * 4;       // 1.5 MB

    // ---- input liquid net: h = tanh(x @ pw1^T + pb1) @ pw2^T + pb2
    launch_gemm(stream, 1, x,   IN,      pw1, pb1, nullptr, xzb, Bn, D / 2, IN);
    launch_gemm(stream, 0, xzb, D / 2,   pw2, pb2, nullptr, h,   Bn, D,     D / 2);

    // ---- 4 mamba blocks (seq len 1)
    for (int l = 0; l < NL; ++l) {
        float* xn = dltb;  // dead before delta is produced
        rmsnorm_kernel<<<Bn, 256, 0, stream>>>(h, norm_w + (size_t)l * D, xn, D);
        launch_gemm(stream, 0, xn, D, in_proj_w + (size_t)l * 2 * DI * D,
                    nullptr, nullptr, xzb, Bn, 2 * DI, D);
        const int tot = Bn * DI;
        convsilu_kernel<<<(tot + 255) / 256, 256, 0, stream>>>(
            xzb, conv_w + (size_t)l * DI * 4, conv_b + (size_t)l * DI, xsb, DI, tot);
        launch_gemm(stream, 0, xsb, DI, x_proj_w + (size_t)l * 96 * DI,
                    nullptr, nullptr, dblb, Bn, 96, DI);
        // delta = softplus(dlo @ dt_w^T + dt_b); dlo = dbl[:, :64] (lda = 96)
        launch_gemm(stream, 2, dblb, 96, dt_w + (size_t)l * DI * DTR,
                    dt_b + (size_t)l * DI, nullptr, dltb, Bn, DI, DTR);
        combine_kernel<<<Bn, 256, 0, stream>>>(
            dblb, dltb, xzb, D_ssm + (size_t)l * DI, xsb, DI);
        // h = h + y @ out_w^T (residual in place; disjoint per-thread RMW)
        launch_gemm(stream, 0, xsb, DI, out_w + (size_t)l * D * DI,
                    nullptr, h, h, Bn, D, DI);
    }

    // ---- output liquid net: tanh applied to both per reference (tanh(_liquid))
    float* hf = xzb;  // B x 2048
    launch_gemm(stream, 1, h,  D,     dw1, db1, nullptr, hf, Bn, 2 * D, D);
    launch_gemm(stream, 1, hf, 2 * D, dw2, db2, nullptr, (float*)d_out, Bn, OUT, 2 * D);
}
